// MambaG2G_27479200760103
// MI455X (gfx1250) — matrix operand order repaired and checked by reading, not compiled, not run on hardware
//
#include <hip/hip_runtime.h>
#include <hip/hip_bf16.h>
#include <math.h>

#define D_MODEL  96
#define D_STATE  6
#define D_CONV   9
#define D_INNER  192
#define DT_RANK  6
#define NPROJ    (DT_RANK + 2*D_STATE)   /* 18 */
#define BATCH    16
#define SEQLEN   4096
#define MTOT     (BATCH*SEQLEN)          /* 65536 */
#define XZ_LD    (2*D_INNER)             /* 384 */

typedef __attribute__((ext_vector_type(16))) _Float16 v16h;
typedef __attribute__((ext_vector_type(8)))  float    v8f;
typedef unsigned int u32x4 __attribute__((ext_vector_type(4)));
typedef int          i32x4 __attribute__((ext_vector_type(4)));
typedef int          i32x8 __attribute__((ext_vector_type(8)));

#if defined(__has_builtin)
#  if __has_builtin(__builtin_amdgcn_tensor_load_to_lds) && __has_builtin(__builtin_amdgcn_s_wait_tensorcnt)
#    define HAVE_TDM 1
#  endif
#endif
#ifndef HAVE_TDM
#  define HAVE_TDM 0
#endif

__device__ __forceinline__ float sigmoidf_(float x){ return 1.0f/(1.0f+__expf(-x)); }
__device__ __forceinline__ float siluf_(float x){ return x*sigmoidf_(x); }
__device__ __forceinline__ float softplusf_(float x){ return (x>20.0f)? x : log1pf(__expf(x)); }
__device__ __forceinline__ float eluf_(float x){ return (x>0.0f)? x : expm1f(x); }

#if HAVE_TDM
typedef __attribute__((address_space(3))) unsigned char lds_byte_t;

// Issue a 2D TENSOR_LOAD_TO_LDS: copy `rows` x `cols` elements (elem size code:
// 0=1B,1=2B,2=4B) from global (row stride `stride_elems`) into contiguous LDS.
// D# packing per CDNA5 ISA. Call from ONE wave.
__device__ __forceinline__ void tdm_load_tile(unsigned int lds_off, const void* gptr,
                                              unsigned int esize_code,
                                              unsigned int rows, unsigned int cols,
                                              unsigned int stride_elems) {
  const unsigned long long ga = (unsigned long long)(uintptr_t)gptr;
  u32x4 g0;
  g0[0] = 1u;                                           // count=1, user mode, no gather
  g0[1] = lds_off;                                      // lds_addr (bytes)
  g0[2] = (unsigned int)(ga & 0xFFFFFFFFu);             // global_addr[31:0]
  g0[3] = (unsigned int)((ga >> 32) & 0x1FFFFFFu)       // global_addr[56:32]
        | (2u << 30);                                   // type = 2 ("image")
  i32x8 g1;
  g1[0] = (int)(esize_code << 16);                      // wg_mask=0, data_size
  g1[1] = (int)((cols & 0xFFFFu) << 16);                // tensor_dim0[15:0]
  g1[2] = (int)(((cols >> 16) & 0xFFFFu))               // tensor_dim0[31:16]
        | (int)((rows & 0xFFFFu) << 16);                // tensor_dim1[15:0]
  g1[3] = (int)((rows >> 16) & 0xFFFFu)                 // tensor_dim1[31:16]
        | (int)((cols & 0xFFFFu) << 16);                // tile_dim0
  g1[4] = (int)(rows & 0xFFFFu);                        // tile_dim1 (tile_dim2=0)
  g1[5] = (int)stride_elems;                            // tensor_dim0_stride[31:0]
  g1[6] = 0;                                            // stride hi / dim1_stride lo
  g1[7] = 0;
  const i32x4 z4 = {0, 0, 0, 0};
#if __clang_major__ >= 23
  const i32x8 z8 = {0, 0, 0, 0, 0, 0, 0, 0};
  __builtin_amdgcn_tensor_load_to_lds(g0, g1, z4, z4, z8, 0);
#else
  __builtin_amdgcn_tensor_load_to_lds(g0, g1, z4, z4, 0);
#endif
}
#define LDS_OFFSET(p) ((unsigned int)(uintptr_t)(lds_byte_t*)(p))
#endif // HAVE_TDM

// ---- WMMA fragment builders (per CDNA5 ISA VGPR layouts, wave32) ----------

// A: 16x32 f16 tile, row-major with stride lda (elements).
__device__ __forceinline__ v16h load_a_frag(const _Float16* A, int lda, int lane) {
  v16h a;
  const int h = lane >> 4;
  const int m = lane & 15;
  const _Float16* row = A + m * lda;
#pragma unroll
  for (int v = 0; v < 8; ++v) {
    const int k = (v < 4) ? (h*8 + 2*v) : (16 + h*8 + 2*(v-4));
    a[2*v]   = row[k];
    a[2*v+1] = row[k+1];
  }
  return a;
}

// B: 32x16 f16 tile where B[k][n] = W[nbase+n][kbase+k], W row-major [N,K] fp32.
__device__ __forceinline__ v16h load_b_fragT(const float* W, int ldw, int nbase, int kbase, int lane) {
  v16h b;
  const int h = lane >> 4;
  const int n = lane & 15;
  const float* wrow = W + (size_t)(nbase + n) * ldw + kbase;
#pragma unroll
  for (int v = 0; v < 8; ++v) {
    const int k = (v < 4) ? (h*8 + 2*v) : (16 + h*8 + 2*(v-4));
    b[2*v]   = (_Float16)wrow[k];
    b[2*v+1] = (_Float16)wrow[k+1];
  }
  return b;
}

// C/D f32 16x16: VGPR r -> M = r + 8*h, N = lane&15
__device__ __forceinline__ void store_d_frag(float* D, int ldd, const v8f c, int lane) {
  const int h = lane >> 4;
  const int n = lane & 15;
#pragma unroll
  for (int r = 0; r < 8; ++r)
    D[(size_t)(r + 8*h) * ldd + n] = c[r];
}

// ---- Kernel 1: RMSNorm + in_proj GEMM (WMMA, x tile staged via TDM) -------
// grid.x = MTOT/16, block = 256 (8 waves); wave w owns N-tiles w*3 .. w*3+2
__global__ __launch_bounds__(256)
void k_rms_inproj(const float* __restrict__ x, const float* __restrict__ in_proj_w,
                  const float* __restrict__ norm_w, float* __restrict__ xz) {
  __shared__ float    xstage[16][D_MODEL];  // raw fp32 x tile (TDM destination)
  __shared__ _Float16 aT[16][104];          // normalized f16 A tile
  const int tid = threadIdx.x;
  const int m0  = blockIdx.x * 16;

#if HAVE_TDM
  if (tid < 32) {   // wave 0 issues the 16x96 fp32 tile DMA (global -> LDS)
    tdm_load_tile(LDS_OFFSET(&xstage[0][0]), x + (size_t)m0 * D_MODEL,
                  /*4B*/2u, 16u, (unsigned)D_MODEL, (unsigned)D_MODEL);
    __builtin_amdgcn_s_wait_tensorcnt((short)0);
  }
  __syncthreads();
#else
  for (int i = tid; i < 16*D_MODEL; i += 256)
    xstage[i / D_MODEL][i % D_MODEL] = x[(size_t)m0 * D_MODEL + i];
  __syncthreads();
#endif

  if (tid < 16) {
    float ss = 0.f;
#pragma unroll 8
    for (int j = 0; j < D_MODEL; ++j) { const float v = xstage[tid][j]; ss += v*v; }
    const float r = rsqrtf(ss * (1.0f/D_MODEL) + 1e-5f);
#pragma unroll 8
    for (int j = 0; j < D_MODEL; ++j)
      aT[tid][j] = (_Float16)(xstage[tid][j] * r * norm_w[j]);
  }
  __syncthreads();

  const int wave = tid >> 5;
  const int lane = tid & 31;
#pragma unroll
  for (int t = 0; t < 3; ++t) {
    const int nbase = (wave*3 + t) * 16;
    v8f c = {};
#pragma unroll
    for (int ks = 0; ks < 3; ++ks) {
      const v16h a = load_a_frag(&aT[0][ks*32], 104, lane);
      const v16h b = load_b_fragT(in_proj_w, D_MODEL, nbase, ks*32, lane);
      c = __builtin_amdgcn_wmma_f32_16x16x32_f16(false, a, false, b, (short)0, c, false, false);
    }
    store_d_frag(xz + (size_t)m0 * XZ_LD + nbase, XZ_LD, c, lane);
  }
}

// ---- Kernel 2: causal depthwise conv (width 9) + SiLU ---------------------
__global__ __launch_bounds__(256)
void k_conv_silu(const float* __restrict__ xz, const float* __restrict__ conv_w,
                 const float* __restrict__ conv_b, float* __restrict__ xc) {
  const int idx = blockIdx.x * blockDim.x + threadIdx.x;
  if (idx >= MTOT * D_INNER) return;
  const int e = idx % D_INNER;
  const int l = (idx / D_INNER) % SEQLEN;
  const int b = idx / (D_INNER * SEQLEN);
  const float* wk = conv_w + e * D_CONV;
  float acc = conv_b[e];
#pragma unroll
  for (int j = 0; j < D_CONV; ++j) {
    const int ll = l - (D_CONV-1) + j;
    if (ll >= 0)
      acc += xz[((size_t)(b*SEQLEN + ll)) * XZ_LD + e] * wk[j];
  }
  xc[idx] = siluf_(acc);
}

// ---- Kernel 3: x_proj (18 outs) + dt_proj + softplus ----------------------
__global__ __launch_bounds__(192)
void k_xproj_dt(const float* __restrict__ xc, const float* __restrict__ x_proj_w,
                const float* __restrict__ dt_w, const float* __restrict__ dt_b,
                float* __restrict__ delta, float* __restrict__ Bm, float* __restrict__ Cm) {
  __shared__ float xcl[D_INNER];
  __shared__ float dbc[NPROJ];
  const int tid = threadIdx.x;
  const size_t pos = blockIdx.x;

  xcl[tid] = xc[pos * D_INNER + tid];
  __syncthreads();

  if (tid < NPROJ) {
    const float* wrow = x_proj_w + tid * D_INNER;
    float s = 0.f;
#pragma unroll 8
    for (int e = 0; e < D_INNER; ++e) s += xcl[e] * wrow[e];
    dbc[tid] = s;
  }
  __syncthreads();

  {
    float d = dt_b[tid];
    const float* wrow = dt_w + tid * DT_RANK;
#pragma unroll
    for (int r = 0; r < DT_RANK; ++r) d += dbc[r] * wrow[r];
    delta[pos * D_INNER + tid] = softplusf_(d);
  }
  if (tid >= DT_RANK && tid < DT_RANK + D_STATE)
    Bm[pos * D_STATE + (tid - DT_RANK)] = dbc[tid];
  else if (tid >= DT_RANK + D_STATE && tid < NPROJ)
    Cm[pos * D_STATE + (tid - DT_RANK - D_STATE)] = dbc[tid];
}

// ---- Kernel 4: selective scan + D skip + z gating (f16 output) ------------
__global__ __launch_bounds__(192)
void k_scan(const float* __restrict__ delta, const float* __restrict__ xc,
            const float* __restrict__ Bm, const float* __restrict__ Cm,
            const float* __restrict__ xz, const float* __restrict__ A_log,
            const float* __restrict__ D_param, _Float16* __restrict__ yg) {
  const int e = threadIdx.x;
  const int b = blockIdx.x;
  __shared__ float Bs[D_STATE], Cs[D_STATE];

  float Ae[D_STATE];
#pragma unroll
  for (int n = 0; n < D_STATE; ++n) Ae[n] = -__expf(A_log[e*D_STATE + n]);
  float h[D_STATE];
#pragma unroll
  for (int n = 0; n < D_STATE; ++n) h[n] = 0.f;
  const float Dp = D_param[e];

  for (int l = 0; l < SEQLEN; ++l) {
    const size_t pos = (size_t)b * SEQLEN + l;
    if (e < D_STATE)        Bs[e]         = Bm[pos*D_STATE + e];
    else if (e < 2*D_STATE) Cs[e-D_STATE] = Cm[pos*D_STATE + (e-D_STATE)];
    __syncthreads();

    const float dl = delta[pos*D_INNER + e];
    const float xv = xc[pos*D_INNER + e];
    const float dx = dl * xv;
    __builtin_prefetch(delta + (pos+1)*D_INNER + e, 0, 1);  // global_prefetch_b8
    __builtin_prefetch(xc    + (pos+1)*D_INNER + e, 0, 1);

    float y = 0.f;
#pragma unroll
    for (int n = 0; n < D_STATE; ++n) {
      const float dA = __expf(dl * Ae[n]);
      h[n] = dA * h[n] + dx * Bs[n];
      y += h[n] * Cs[n];
    }
    y += Dp * xv;
    const float z = xz[pos*XZ_LD + D_INNER + e];
    yg[pos*D_INNER + e] = (_Float16)(y * siluf_(z));
    __syncthreads();
  }
}

// ---- Kernel 5: out_proj GEMM (WMMA) + residual; A tile via TDM ------------
// out[m,d] = x[m,d] + yg[m,:] @ out_proj_w^T ; grid = MTOT/16, block = 192 (6 waves)
__global__ __launch_bounds__(192)
void k_outproj(const _Float16* __restrict__ yg, const float* __restrict__ out_proj_w,
               const float* __restrict__ x, float* __restrict__ outb) {
  __shared__ _Float16 aT[16][D_INNER];   // contiguous: matches TDM 2D copy
  const int tid = threadIdx.x;
  const int m0  = blockIdx.x * 16;

#if HAVE_TDM
  if (tid < 32) {   // wave 0: DMA the 16x192 f16 tile (384B rows) into LDS
    tdm_load_tile(LDS_OFFSET(&aT[0][0]), yg + (size_t)m0 * D_INNER,
                  /*2B*/1u, 16u, (unsigned)D_INNER, (unsigned)D_INNER);
    __builtin_amdgcn_s_wait_tensorcnt((short)0);
  }
  __syncthreads();
#else
  for (int i = tid; i < 16*D_INNER; i += 192)
    aT[i / D_INNER][i % D_INNER] = yg[(size_t)m0 * D_INNER + i];
  __syncthreads();
#endif

  const int wave = tid >> 5;          // 0..5 -> one 16-wide N tile each (96 = 6*16)
  const int lane = tid & 31;
  const int nbase = wave * 16;
  v8f c = {};
#pragma unroll
  for (int ks = 0; ks < 6; ++ks) {
    const v16h a = load_a_frag(&aT[0][ks*32], D_INNER, lane);
    const v16h b = load_b_fragT(out_proj_w, D_INNER, nbase, ks*32, lane);
    c = __builtin_amdgcn_wmma_f32_16x16x32_f16(false, a, false, b, (short)0, c, false, false);
  }
  const int h = lane >> 4, n = lane & 15;
#pragma unroll
  for (int r = 0; r < 8; ++r) {
    const size_t row = (size_t)(m0 + r + 8*h);
    outb[row*D_MODEL + nbase + n] = c[r] + x[row*D_MODEL + nbase + n];
  }
}

// ---- Kernel 6: mean over L -------------------------------------------------
__global__ __launch_bounds__(256)
void k_mean(const float* __restrict__ outb, float* __restrict__ eacc) {
  __shared__ float red[256];
  const int b = blockIdx.x / D_MODEL;
  const int d = blockIdx.x % D_MODEL;
  const int tid = threadIdx.x;
  float s = 0.f;
  for (int l = tid; l < SEQLEN; l += 256)
    s += outb[((size_t)(b*SEQLEN + l))*D_MODEL + d];
  red[tid] = s;
  __syncthreads();
  for (int off = 128; off > 0; off >>= 1) {
    if (tid < off) red[tid] += red[tid + off];
    __syncthreads();
  }
  if (tid == 0) eacc[b*D_MODEL + d] = red[0] * (1.0f/SEQLEN);
}

// ---- Kernel 7: head (fc -> tanh -> elu -> mu/sigma) -----------------------
__global__ __launch_bounds__(64)
void k_head(const float* __restrict__ eacc,
            const float* __restrict__ fc_w, const float* __restrict__ fc_b,
            const float* __restrict__ mu_w, const float* __restrict__ mu_b,
            const float* __restrict__ sg_w, const float* __restrict__ sg_b,
            float* __restrict__ out) {
  __shared__ float ev[D_MODEL];
  __shared__ float ft[64];
  const int b = blockIdx.x;
  const int tid = threadIdx.x;
  for (int i = tid; i < D_MODEL; i += 64) ev[i] = eacc[b*D_MODEL + i];
  __syncthreads();

  {
    float s = fc_b[tid];
    const float* w = fc_w + tid * D_MODEL;
#pragma unroll 8
    for (int d = 0; d < D_MODEL; ++d) s += ev[d] * w[d];
    const float f = eluf_(tanhf(s));
    ft[tid] = f;
    out[b*64 + tid] = f;
  }
  __syncthreads();

  {
    float m = mu_b[tid], g = sg_b[tid];
    const float* wm = mu_w + tid * 64;
    const float* wg = sg_w + tid * 64;
#pragma unroll 8
    for (int i = 0; i < 64; ++i) { m += ft[i]*wm[i]; g += ft[i]*wg[i]; }
    out[16*64   + b*64 + tid] = m;
    out[2*16*64 + b*64 + tid] = eluf_(g) + 1.0f + 1e-14f;
  }
}

// ---- Launch ----------------------------------------------------------------
extern "C" void kernel_launch(void* const* d_in, const int* in_sizes, int n_in,
                              void* d_out, int out_size, void* d_ws, size_t ws_size,
                              hipStream_t stream) {
  const float* x         = (const float*)d_in[0];
  const float* in_proj_w = (const float*)d_in[1];
  const float* conv_w    = (const float*)d_in[2];
  const float* conv_b    = (const float*)d_in[3];
  const float* x_proj_w  = (const float*)d_in[4];
  const float* dt_proj_w = (const float*)d_in[5];
  const float* dt_proj_b = (const float*)d_in[6];
  const float* A_log     = (const float*)d_in[7];
  const float* D_param   = (const float*)d_in[8];
  const float* out_proj_w= (const float*)d_in[9];
  const float* norm_w    = (const float*)d_in[10];
  const float* out_fc_w  = (const float*)d_in[11];
  const float* out_fc_b  = (const float*)d_in[12];
  const float* mu_w      = (const float*)d_in[13];
  const float* mu_b      = (const float*)d_in[14];
  const float* sigma_w   = (const float*)d_in[15];
  const float* sigma_b   = (const float*)d_in[16];
  float* out = (float*)d_out;

  char* p = (char*)d_ws;
  float*     xz    = (float*)p;     p += (size_t)MTOT * XZ_LD   * sizeof(float);
  float*     xc    = (float*)p;     p += (size_t)MTOT * D_INNER * sizeof(float);
  float*     delta = (float*)p;     p += (size_t)MTOT * D_INNER * sizeof(float);
  float*     Bm    = (float*)p;     p += (size_t)MTOT * D_STATE * sizeof(float);
  float*     Cm    = (float*)p;     p += (size_t)MTOT * D_STATE * sizeof(float);
  _Float16*  yg    = (_Float16*)p;  p += (size_t)MTOT * D_INNER * sizeof(_Float16);
  float*     outb  = (float*)p;     p += (size_t)MTOT * D_MODEL * sizeof(float);
  float*     eacc  = (float*)p;     p += (size_t)BATCH * D_MODEL * sizeof(float);

  k_rms_inproj<<<MTOT/16, 256, 0, stream>>>(x, in_proj_w, norm_w, xz);
  k_conv_silu <<<(MTOT*D_INNER + 255)/256, 256, 0, stream>>>(xz, conv_w, conv_b, xc);
  k_xproj_dt  <<<MTOT, D_INNER, 0, stream>>>(xc, x_proj_w, dt_proj_w, dt_proj_b, delta, Bm, Cm);
  k_scan      <<<BATCH, D_INNER, 0, stream>>>(delta, xc, Bm, Cm, xz, A_log, D_param, yg);
  k_outproj   <<<MTOT/16, 192, 0, stream>>>(yg, out_proj_w, x, outb);
  k_mean      <<<BATCH*D_MODEL, 256, 0, stream>>>(outb, eacc);
  k_head      <<<BATCH, 64, 0, stream>>>(eacc, out_fc_w, out_fc_b, mu_w, mu_b, sigma_w, sigma_b, out);
}
